// GeneralAttention_49091476193657
// MI455X (gfx1250) — matrix operand order repaired and checked by reading, not compiled, not run on hardware
//
#include <hip/hip_runtime.h>
#include <hip/hip_bf16.h>

// Windowed attention + output projection for MI455X (gfx1250, wave32, WMMA).
// Precision: bf16 inputs to v_wmma_f32_16x16x32_bf16, f32 accumulation.
//
// ws layout (ushort units):
//   [ qbf : B*T*E = 4M ][ kT : B*H*D*T = 4M ][ vbf : 4M ][ ybf : 4M ][ Wt : 1M ]
//   total 17M ushorts = 34 MB.

#define BB 2
#define TT 2048
#define EE 1024
#define HH 16
#define DD 64
#define WWIN 64

typedef __attribute__((ext_vector_type(16))) __bf16 v16bf;
typedef __attribute__((ext_vector_type(8)))  float  v8f;

union F8   { v8f   v; float          f[8];  };
union AB16 { v16bf v; unsigned short u[16]; uint4 q2[2]; };

__device__ __forceinline__ unsigned short f2bf(float x) {
    union { float f; unsigned u; } c; c.f = x;
    unsigned u = c.u;
    u += 0x7FFFu + ((u >> 16) & 1u);   // round-to-nearest-even
    return (unsigned short)(u >> 16);
}

__device__ __forceinline__ int iclamp(int x, int lo, int hi) {
    return x < lo ? lo : (x > hi ? hi : x);       // -> v_med3_i32
}

// ---------------------------------------------------------------------------
// Kernel 0: convert q,v -> bf16 (same layout); k -> bf16 split per head:
//           kT[((b*H + h)*T + t)*D + d] = k[b,t,h*D+d]
// ---------------------------------------------------------------------------
__global__ __launch_bounds__(256)
void prep_kernel(const float* __restrict__ q, const float* __restrict__ k,
                 const float* __restrict__ v,
                 unsigned short* __restrict__ qbf,
                 unsigned short* __restrict__ kT,
                 unsigned short* __restrict__ vbf) {
    unsigned idx = blockIdx.x * 256u + threadIdx.x;   // 0 .. B*T*E-1
    unsigned e = idx & (EE - 1);
    unsigned t = (idx >> 10) & (TT - 1);
    unsigned b = idx >> 21;
    unsigned h = e >> 6, d = e & 63;
    qbf[idx] = f2bf(q[idx]);
    vbf[idx] = f2bf(v[idx]);
    kT[((size_t)(b * HH + h) * TT + t) * DD + d] = f2bf(k[idx]);
}

// ---------------------------------------------------------------------------
// Kernel 1: windowed flash-style attention, one wave32 per (b, h, 16-query tile)
// ---------------------------------------------------------------------------
__global__ __launch_bounds__(32)
void attn_kernel(const unsigned short* __restrict__ qbf,
                 const unsigned short* __restrict__ kT,
                 const unsigned short* __restrict__ vbf,
                 unsigned short* __restrict__ ybf) {
    const int lane = threadIdx.x;     // 0..31
    const int half = lane >> 4;       // A/C layout half-wave
    const int m    = lane & 15;
    const int qt   = blockIdx.x;      // query tile (T/16 = 128)
    const int h    = blockIdx.y;
    const int b    = blockIdx.z;
    const int q0   = qt * 16;
    const int kt0  = q0 - 64;         // first candidate key (16-aligned, may be <0)

    const unsigned short* qb  = qbf + (size_t)(b * TT) * EE + h * DD;
    const unsigned short* vB  = vbf + (size_t)(b * TT) * EE + h * DD;
    const unsigned short* kTb = kT  + (size_t)(b * HH + h) * DD * TT;

    __shared__ unsigned short Pl[16][160];   // probabilities, bf16, [row][key]

    // ---- Q A-fragments: 16x32 bf16, head-dim chunks [0,32) and [32,64)
    // A layout: lane -> row m, K-runs are contiguous 8-spans -> b128 loads
    AB16 AQ[2];
    {
        const unsigned short* qrow = qb + (size_t)(q0 + m) * EE + half * 8;
        AQ[0].q2[0] = *(const uint4*)(qrow);            // K  0+half*8 ..
        AQ[0].q2[1] = *(const uint4*)(qrow + 16);       // K 16+half*8 ..
        AQ[1].q2[0] = *(const uint4*)(qrow + 32);
        AQ[1].q2[1] = *(const uint4*)(qrow + 48);
    }

    // ---- scores: 10 fragments of 16 queries x 16 keys (keys kt0 .. kt0+159)
    // Each 16-key fragment is fully in-range or fully masked -> clamp base.
    F8 S[10];
#pragma unroll
    for (int f = 0; f < 10; ++f) {
        const int kbase = kt0 + 16 * f;
        const int cb    = iclamp(kbase, 0, TT - 16);
        // B = K^T: lane -> key col, element -> d row (8-spans); kT rows are contiguous
        const unsigned short* krow0 = kTb + (size_t)(cb + m) * DD + half * 8;
        const unsigned short* krow1 = kTb + (size_t)(cb + m) * DD + 32 + half * 8;
        AB16 BK0, BK1;
        BK0.q2[0] = *(const uint4*)(krow0);
        BK0.q2[1] = *(const uint4*)(krow0 + 16);
        BK1.q2[0] = *(const uint4*)(krow1);
        BK1.q2[1] = *(const uint4*)(krow1 + 16);
        v8f z; F8* zz = (F8*)&z;
#pragma unroll
        for (int i = 0; i < 8; ++i) zz->f[i] = 0.0f;
        z = __builtin_amdgcn_wmma_f32_16x16x32_bf16(false, AQ[0].v, false, BK0.v,
                                                    (short)0, z, false, false);
        S[f].v = __builtin_amdgcn_wmma_f32_16x16x32_bf16(false, AQ[1].v, false, BK1.v,
                                                         (short)0, z, false, false);
    }

    // ---- scale + window/bounds mask + row softmax
    const float scale = 0.125f;   // 1/sqrt(64)
    float mrow[8], lrow[8];
#pragma unroll
    for (int r = 0; r < 8; ++r) mrow[r] = -1e30f;

#pragma unroll
    for (int f = 0; f < 10; ++f) {
        int j = kt0 + 16 * f + m;                  // this lane's key column
#pragma unroll
        for (int r = 0; r < 8; ++r) {
            int i = q0 + half * 8 + r;             // this vgpr's query row
            int dist = i - j; dist = dist < 0 ? -dist : dist;
            bool ok = (j >= 0) && (j < TT) && (dist <= WWIN);
            float s = ok ? S[f].f[r] * scale : -1e30f;
            S[f].f[r] = s;
            mrow[r] = fmaxf(mrow[r], s);
        }
    }
#pragma unroll
    for (int r = 0; r < 8; ++r) {                  // row max across 16 lanes
        float mv = mrow[r];
        mv = fmaxf(mv, __shfl_xor(mv, 1, 32));
        mv = fmaxf(mv, __shfl_xor(mv, 2, 32));
        mv = fmaxf(mv, __shfl_xor(mv, 4, 32));
        mv = fmaxf(mv, __shfl_xor(mv, 8, 32));
        mrow[r] = mv;
        lrow[r] = 0.0f;
    }
#pragma unroll
    for (int f = 0; f < 10; ++f) {
#pragma unroll
        for (int r = 0; r < 8; ++r) {
            float p = __expf(S[f].f[r] - mrow[r]); // masked -> exp(-1e30) == 0
            lrow[r] += p;
            Pl[half * 8 + r][16 * f + m] = f2bf(p);
        }
    }
#pragma unroll
    for (int r = 0; r < 8; ++r) {                  // row sum across 16 lanes
        float lv = lrow[r];
        lv += __shfl_xor(lv, 1, 32);
        lv += __shfl_xor(lv, 2, 32);
        lv += __shfl_xor(lv, 4, 32);
        lv += __shfl_xor(lv, 8, 32);
        lrow[r] = lv;
    }
    __syncthreads();

    // ---- O = P x V ; 5 key chunks of 32, 4 output-dim chunks of 16
    F8 O[4];
#pragma unroll
    for (int nb = 0; nb < 4; ++nb)
#pragma unroll
        for (int i = 0; i < 8; ++i) O[nb].f[i] = 0.0f;

#pragma unroll
    for (int c = 0; c < 5; ++c) {
        AB16 AP;                                   // P as A-frag from LDS (b128 x2)
        AP.q2[0] = *(const uint4*)&Pl[m][32 * c + half * 8];
        AP.q2[1] = *(const uint4*)&Pl[m][32 * c + 16 + half * 8];

#pragma unroll
        for (int nb = 0; nb < 4; ++nb) {
            AB16 BV;
#pragma unroll
            for (int s = 0; s < 16; ++s) {
                int ck = iclamp(kt0 + 32 * c + (s < 8 ? half * 8 + s : 16 + half * 8 + (s - 8)), 0, TT - 1);
                BV.u[s] = vB[(size_t)ck * EE + nb * 16 + m];
            }
            O[nb].v = __builtin_amdgcn_wmma_f32_16x16x32_bf16(false, AP.v, false, BV.v,
                                                              (short)0, O[nb].v, false, false);
        }
    }

    // ---- normalize and store y (bf16) to workspace
#pragma unroll
    for (int r = 0; r < 8; ++r) {
        float inv = 1.0f / lrow[r];
        size_t rbase = (size_t)(b * TT + q0 + half * 8 + r) * EE + h * DD;
#pragma unroll
        for (int nb = 0; nb < 4; ++nb)
            ybf[rbase + nb * 16 + m] = f2bf(O[nb].f[r] * inv);
    }
}

// ---------------------------------------------------------------------------
// Kernel 2: W (f32, [o][k]) -> Wt (bf16, [o][k]) so GEMM B-frags are contiguous
// ---------------------------------------------------------------------------
__global__ __launch_bounds__(256)
void wt_kernel(const float* __restrict__ W, unsigned short* __restrict__ Wt) {
    int idx = blockIdx.x * 256 + threadIdx.x;      // coalesced read of W
    int kk  = idx & (EE - 1);
    int o   = idx >> 10;
    Wt[(size_t)o * EE + kk] = f2bf(W[idx]);
}

// ---------------------------------------------------------------------------
// Kernel 3: out = y @ W^T + b ; one wave32 per 16x64 output tile, 128 WMMAs
// ---------------------------------------------------------------------------
__global__ __launch_bounds__(32)
void proj_kernel(const unsigned short* __restrict__ ybf,
                 const unsigned short* __restrict__ Wt,
                 const float* __restrict__ bias,
                 float* __restrict__ out) {
    const int lane = threadIdx.x;
    const int half = lane >> 4;
    const int m    = lane & 15;
    const int ob   = blockIdx.x * 64;              // output-dim tile
    const int mt   = blockIdx.y;                   // row tile (4096/16 = 256)

    F8 acc[4];
#pragma unroll
    for (int nb = 0; nb < 4; ++nb)
#pragma unroll
        for (int i = 0; i < 8; ++i) acc[nb].f[i] = 0.0f;

    const unsigned short* arow = ybf + (size_t)(mt * 16 + m) * EE + half * 8;
    for (int c = 0; c < EE / 32; ++c) {
        AB16 A;                                    // two contiguous 8-bf16 runs
        A.q2[0] = *(const uint4*)(arow + 32 * c);
        A.q2[1] = *(const uint4*)(arow + 32 * c + 16);

        const unsigned short* wrow = Wt + (size_t)(ob + m) * EE + 32 * c + half * 8;
#pragma unroll
        for (int nb = 0; nb < 4; ++nb) {
            AB16 Bf;                               // lane -> N col, element -> K row
            Bf.q2[0] = *(const uint4*)(wrow + (size_t)nb * 16 * EE);
            Bf.q2[1] = *(const uint4*)(wrow + (size_t)nb * 16 * EE + 16);
            acc[nb].v = __builtin_amdgcn_wmma_f32_16x16x32_bf16(false, A.v, false, Bf.v,
                                                                (short)0, acc[nb].v, false, false);
        }
    }

#pragma unroll
    for (int nb = 0; nb < 4; ++nb) {
#pragma unroll
        for (int r = 0; r < 8; ++r) {
            int o = ob + nb * 16 + m;
            out[(size_t)(mt * 16 + half * 8 + r) * EE + o] = acc[nb].f[r] + bias[o];
        }
    }
}

// ---------------------------------------------------------------------------
extern "C" void kernel_launch(void* const* d_in, const int* in_sizes, int n_in,
                              void* d_out, int out_size, void* d_ws, size_t ws_size,
                              hipStream_t stream) {
    const float* q    = (const float*)d_in[0];
    const float* k    = (const float*)d_in[1];
    const float* v    = (const float*)d_in[2];
    const float* W    = (const float*)d_in[3];
    const float* bias = (const float*)d_in[4];
    float* out        = (float*)d_out;

    const size_t NQKV = (size_t)BB * TT * EE;      // 4M elements
    unsigned short* qbf = (unsigned short*)d_ws;
    unsigned short* kT  = qbf + NQKV;
    unsigned short* vbf = kT  + NQKV;
    unsigned short* ybf = vbf + NQKV;
    unsigned short* Wt  = ybf + NQKV;

    prep_kernel<<<dim3(NQKV / 256), dim3(256), 0, stream>>>(q, k, v, qbf, kT, vbf);
    wt_kernel  <<<dim3((EE * EE) / 256), dim3(256), 0, stream>>>(W, Wt);
    attn_kernel<<<dim3(TT / 16, HH, BB), dim3(32), 0, stream>>>(qbf, kT, vbf, ybf);
    proj_kernel<<<dim3(EE / 64, (BB * TT) / 16), dim3(32), 0, stream>>>(ybf, Wt, bias, out);
}
